// RDrop_43628277793360
// MI455X (gfx1250) — compile-verified, matrix operand order checked by reading, not run on hardware
//
#include <hip/hip_runtime.h>

// GCN 2-layer forward for MI455X (gfx1250, wave32, WMMA).
//   Bp1/Bp2 = W1/W2 pre-packed to bf16 WMMA B-fragment layout (one-time, L2-resident)
//   h1   = x @ W1                      (bf16 WMMA, f32 accum)
//   agg1 = b1 + scatter_add(h1[src]*ew -> dst)   (f32 atomics, L2-resident)
//   h2   = relu(agg1) @ W2             (bf16 WMMA, relu fused into A load)
//   out  = b2 + scatter_add(h2[src]*ew -> dst)

#define FIN 512
#define HID 64

typedef __attribute__((ext_vector_type(16))) __bf16 v16bf;
typedef __attribute__((ext_vector_type(8)))  float  v8f;

// ---- A fragment: 16x32 bf16 tile of row-major fp32 matrix X [nrows x ld] ----
// ISA layout (16-bit A 16x32): lane l<16 holds row r=l, K = k0+{0..7, 16..23};
// lanes 16..31 hold row r=l-16, K = k0+{8..15, 24..31}.
template <bool RELU>
__device__ inline v16bf load_a_frag(const float* __restrict__ X, int ld,
                                    int row, int k0, int half) {
  const float* p0 = X + (size_t)row * ld + k0 + half * 8;
  const float* p1 = p0 + 16;
  __builtin_prefetch(p0 + 32, 0, 3);   // next k-tile of this row -> global_prefetch_b8
  v16bf a;
#pragma unroll
  for (int j = 0; j < 8; ++j) {
    float f0 = p0[j];
    float f1 = p1[j];
    if (RELU) { f0 = fmaxf(f0, 0.0f); f1 = fmaxf(f1, 0.0f); }
    a[j]     = (__bf16)f0;
    a[j + 8] = (__bf16)f1;
  }
  return a;
}

// ---- One-time pack: W [K x 64] fp32 row-major -> bf16 B-fragments --------------
// Fragment (kt, nt) covers K rows kt*32..+31, cols nt*16..+15. Per-lane 16 bf16
// stored contiguously (32 B/lane, 1 KB/fragment) so the GEMM loads each fragment
// with a single 32-byte vector load.
// Lane l<16 holds col c = nt*16+l, K = k0+{0..7,16..23}; lanes 16..31 K = {8..15,24..31}.
__global__ __launch_bounds__(128) void pack_b(const float* __restrict__ W,
                                              __bf16* __restrict__ Bp, int K) {
  const int nfrags = (K >> 5) * 4;
  const int frag = blockIdx.x * (blockDim.x >> 5) + (threadIdx.x >> 5);
  if (frag >= nfrags) return;
  const int kt = frag >> 2;
  const int nt = frag & 3;
  const int lane = threadIdx.x & 31;
  const int half = lane >> 4;
  const int r    = lane & 15;
  const int c  = nt * 16 + r;
  const int k0 = kt * 32;
  v16bf b;
#pragma unroll
  for (int j = 0; j < 8; ++j) {
    b[j]     = (__bf16)W[(size_t)(k0 + half * 8 + j) * HID + c];
    b[j + 8] = (__bf16)W[(size_t)(k0 + 16 + half * 8 + j) * HID + c];
  }
  *(v16bf*)(Bp + ((size_t)frag * 32 + lane) * 16) = b;
}

// One wave computes a 16-row x 64-col output tile: 4 accumulators of 16x16 f32.
// B comes pre-packed: fragment (kt,nt) at Bp[(kt*4+nt)*32 + lane], one v16bf load.
template <int K, bool RELU>
__global__ __launch_bounds__(128) void gemm_wmma(const float* __restrict__ X,
                                                 const v16bf* __restrict__ Bp,
                                                 float* __restrict__ H,
                                                 int nrows) {
  const int ntiles = (nrows + 15) >> 4;
  const int wave = blockIdx.x * (blockDim.x >> 5) + (threadIdx.x >> 5);
  if (wave >= ntiles) return;          // wave-uniform exit; EXEC stays all-ones
  const int lane = threadIdx.x & 31;
  const int half = lane >> 4;
  const int r    = lane & 15;
  int row = wave * 16 + r;
  row = row < nrows ? row : nrows - 1; // N=100000 is 16-divisible; clamp is insurance

  v8f acc[4] = {{}, {}, {}, {}};
  for (int kt = 0; kt < (K >> 5); ++kt) {
    v16bf a = load_a_frag<RELU>(X, K, row, kt * 32, half);
#pragma unroll
    for (int nt = 0; nt < 4; ++nt) {
      v16bf b = Bp[((size_t)kt * 4 + nt) * 32 + lane];
      acc[nt] = __builtin_amdgcn_wmma_f32_16x16x32_bf16(
          /*neg_a=*/false, a, /*neg_b=*/false, b,
          /*c_mod=*/(short)0, acc[nt], /*reuse_a=*/false, /*reuse_b=*/false);
    }
  }

  // C/D layout: VGPR v, lanes 0-15 -> (M = v,   N = lane);
  //                     lanes 16-31 -> (M = 8+v, N = lane-16)
  const int mbase = wave * 16 + half * 8;
#pragma unroll
  for (int v = 0; v < 8; ++v) {
    const int m = mbase + v;
    if (m < nrows) {
#pragma unroll
      for (int nt = 0; nt < 4; ++nt)
        H[(size_t)m * HID + nt * 16 + r] = acc[nt][v];
    }
  }
}

// One wave per edge: coalesced 256B gather of H[src], scale by ew, 2 f32 atomics
// per lane into agg[dst]. agg is 25.6 MB -> L2-resident, atomics run in L2.
__global__ __launch_bounds__(256) void edge_scatter(const float* __restrict__ H,
                                                    const int* __restrict__ src,
                                                    const int* __restrict__ dst,
                                                    const float* __restrict__ ew,
                                                    float* __restrict__ agg,
                                                    int nedges) {
  const int e = blockIdx.x * (blockDim.x >> 5) + (threadIdx.x >> 5);
  if (e >= nedges) return;
  const int lane = threadIdx.x & 31;
  const int s = src[e];
  const int d = dst[e];
  const float w = ew[e];
  const float* hp = H + (size_t)s * HID;
  float* ap = agg + (size_t)d * HID;
  const float v0 = hp[lane] * w;
  const float v1 = hp[lane + 32] * w;
  atomicAdd(ap + lane, v0);
  atomicAdd(ap + lane + 32, v1);
}

// Seed aggregation targets with their biases (bias is added post-aggregation).
__global__ __launch_bounds__(256) void init_bias(float* __restrict__ agg1,
                                                 float* __restrict__ out,
                                                 const float* __restrict__ b1,
                                                 const float* __restrict__ b2,
                                                 int total) {
  const int i = blockIdx.x * blockDim.x + threadIdx.x;
  if (i < total) {
    const int f = i & (HID - 1);
    agg1[i] = b1[f];
    out[i]  = b2[f];
  }
}

extern "C" void kernel_launch(void* const* d_in, const int* in_sizes, int n_in,
                              void* d_out, int out_size, void* d_ws, size_t ws_size,
                              hipStream_t stream) {
  const float* x  = (const float*)d_in[0];   // [N, 512]
  const int*   ei = (const int*)d_in[1];     // [2, E] (int32 under JAX default x64-off)
  const float* ew = (const float*)d_in[2];   // [E]
  const float* W1 = (const float*)d_in[3];   // [512, 64]
  const float* b1 = (const float*)d_in[4];   // [64]
  const float* W2 = (const float*)d_in[5];   // [64, 64]
  const float* b2 = (const float*)d_in[6];   // [64]
  float* out = (float*)d_out;                // [N, 64]

  const int Nn = in_sizes[0] / FIN;
  const int Ee = in_sizes[2];
  const int* src = ei;
  const int* dst = ei + Ee;

  const size_t per = (size_t)Nn * HID * sizeof(float);   // 25.6 MB, 32B-aligned
  float*  h1   = (float*)d_ws;
  float*  agg1 = (float*)((char*)d_ws + per);
  float*  h2   = (float*)((char*)d_ws + 2 * per);
  __bf16* Bp1  = (__bf16*)((char*)d_ws + 3 * per);                  // 64 KB
  __bf16* Bp2  = (__bf16*)((char*)d_ws + 3 * per + 64 * 1024);     //  8 KB

  // One-time weight packing (L2-resident afterwards).
  pack_b<<<(((FIN >> 5) * 4) + 3) / 4, 128, 0, stream>>>(W1, Bp1, FIN);
  pack_b<<<(((HID >> 5) * 4) + 3) / 4, 128, 0, stream>>>(W2, Bp2, HID);

  const int total = Nn * HID;
  init_bias<<<(total + 255) / 256, 256, 0, stream>>>(agg1, out, b1, b2, total);

  const int ntiles = (Nn + 15) / 16;
  const int gblocks = (ntiles + 3) / 4;            // 4 waves (128 threads) per block
  gemm_wmma<FIN, false><<<gblocks, 128, 0, stream>>>(x, (const v16bf*)Bp1, h1, Nn);

  const int sblocks = (Ee + 7) / 8;                // 8 edges (waves) per 256-thread block
  edge_scatter<<<sblocks, 256, 0, stream>>>(h1, src, dst, ew, agg1, Ee);

  gemm_wmma<HID, true><<<gblocks, 128, 0, stream>>>(agg1, (const v16bf*)Bp2, h2, Nn);

  edge_scatter<<<sblocks, 256, 0, stream>>>(h2, src, dst, ew, out, Ee);
}
